// Head_37752762532614
// MI455X (gfx1250) — matrix operand order repaired and checked by reading, not compiled, not run on hardware
//
#include <hip/hip_runtime.h>
#include <hip/hip_bf16.h>
#include <stdint.h>

// B=32, T=1024, C=256, H=64 ; SCALE = 256^-0.666
#define B_ 32
#define T_ 1024
#define C_ 256
#define H_ 64
#define SCALE_F 0.0248950f

typedef __attribute__((ext_vector_type(16))) __bf16 v16bf;
typedef __attribute__((ext_vector_type(8)))  float  v8f;
typedef __attribute__((ext_vector_type(2)))  int    v2i;
typedef __attribute__((ext_vector_type(4)))  int    v4i;

union Frag {
    v16bf v;
    uint4 q[2];
};

static __device__ inline v8f wmma_bf16(v16bf a, v16bf b, v8f c) {
    // D = A(16x32 bf16) * B(32x16 bf16) + C(16x16 f32)
    return __builtin_amdgcn_wmma_f32_16x16x32_bf16(
        false, a, false, b, (short)0, c, false, false);
}

static __device__ inline v8f v8f_zero() {
    v8f z;
#pragma unroll
    for (int i = 0; i < 8; ++i) z[i] = 0.f;
    return z;
}

// 16-lane half-wave reductions: one WMMA C-layout row per (reg, half).
static __device__ inline float rowmax16(float v) {
    v = fmaxf(v, __shfl_xor(v, 1, 32));
    v = fmaxf(v, __shfl_xor(v, 2, 32));
    v = fmaxf(v, __shfl_xor(v, 4, 32));
    v = fmaxf(v, __shfl_xor(v, 8, 32));
    return v;
}
static __device__ inline float rowsum16(float v) {
    v += __shfl_xor(v, 1, 32);
    v += __shfl_xor(v, 2, 32);
    v += __shfl_xor(v, 4, 32);
    v += __shfl_xor(v, 8, 32);
    return v;
}

// ---- CDNA5 async global->LDS copy (ASYNCcnt path), with sync fallback ------
#if defined(__has_builtin)
#if __has_builtin(__builtin_amdgcn_global_load_async_to_lds_b128)
#define ASYNC_B128 1
#elif __has_builtin(__builtin_amdgcn_global_load_async_to_lds_b64)
#define ASYNC_B64 1
#endif
#endif

#if defined(ASYNC_B128)
#define CHUNK_ELEMS 8            // 16B per async op
#define OPS_PER_BLOCK 4          // (2 K + 2 V) per thread per 32-key block
static __device__ inline void async_copy_chunk(const __bf16* g, __bf16* l) {
    __builtin_amdgcn_global_load_async_to_lds_b128(
        (__attribute__((address_space(1))) v4i*)(__bf16*)g,
        (__attribute__((address_space(3))) v4i*)l, 0, 0);
}
#elif defined(ASYNC_B64)
#define CHUNK_ELEMS 4
#define OPS_PER_BLOCK 8
static __device__ inline void async_copy_chunk(const __bf16* g, __bf16* l) {
    __builtin_amdgcn_global_load_async_to_lds_b64(
        (__attribute__((address_space(1))) v2i*)(__bf16*)g,
        (__attribute__((address_space(3))) v2i*)l, 0, 0);
}
#else
#define CHUNK_ELEMS 8
#define OPS_PER_BLOCK 4
static __device__ inline void async_copy_chunk(const __bf16* g, __bf16* l) {
    *(uint4*)l = *(const uint4*)g;   // sync fallback
}
#endif

template <int N>
static __device__ inline void wait_asynccnt() {
#if defined(__has_builtin) && __has_builtin(__builtin_amdgcn_s_wait_asynccnt)
    __builtin_amdgcn_s_wait_asynccnt((short)N);
#else
    asm volatile("s_wait_asynccnt %0" ::"i"(N) : "memory");
#endif
}

// ---------------------------------------------------------------------------
// Kernel 1: q = x*Wq, k = x*Wk, v = x*Wv  (fp32 -> bf16), v stored transposed.
// 256 threads = 8 waves; each wave owns 16 rows of x; block owns 128 rows.
// ---------------------------------------------------------------------------
__global__ __launch_bounds__(256) void proj_qkv(
    const float* __restrict__ x,
    const float* __restrict__ Wq, const float* __restrict__ Wk,
    const float* __restrict__ Wv,
    __bf16* __restrict__ qb, __bf16* __restrict__ kb, __bf16* __restrict__ vtb)
{
    constexpr int WTS = 264;            // padded stride: 264*2B keeps 16B alignment
    __shared__ __bf16 wt[H_ * WTS];     // W^T (h-major), bf16

    const int tid  = threadIdx.x;
    const int lane = tid & 31;
    const int wave = tid >> 5;
    const int l15  = lane & 15;
    const int lh   = lane >> 4;
    const int rowbase = blockIdx.x * 128 + wave * 16;
    const float* xrow = x + (size_t)(rowbase + l15) * C_;

    const float* Ws[3] = {Wq, Wk, Wv};

#pragma unroll                         // w compile-time: straight-line epilogues
    for (int w = 0; w < 3; ++w) {
        __syncthreads();
        const float* W = Ws[w];
        for (int i = tid; i < C_ * H_; i += 256) {  // coalesced read of W[c][h]
            int c = i >> 6, h = i & 63;
            wt[h * WTS + c] = (__bf16)W[i];
        }
        __syncthreads();

        v8f acc[4];
#pragma unroll
        for (int nt = 0; nt < 4; ++nt) acc[nt] = v8f_zero();

#pragma unroll 2
        for (int kk = 0; kk < C_; kk += 32) {
            // A fragment from fp32 x (contiguous K: half-wave owns 16 consecutive K)
            Frag a;
            const float4* pA0 = (const float4*)(xrow + kk + lh * 16);
            const float4* pA1 = (const float4*)(xrow + kk + lh * 16 + 8);
            float4 f0 = pA0[0], f1 = pA0[1], f2 = pA1[0], f3 = pA1[1];
            float fa[8] = {f0.x, f0.y, f0.z, f0.w, f1.x, f1.y, f1.z, f1.w};
            float fb[8] = {f2.x, f2.y, f2.z, f2.w, f3.x, f3.y, f3.z, f3.w};
#pragma unroll
            for (int e = 0; e < 8; ++e) {
                a.v[e]     = (__bf16)fa[e];
                a.v[e + 8] = (__bf16)fb[e];
            }
            // batch-preload all 4 B fragments, then back-to-back WMMAs
            Frag bfr[4];
#pragma unroll
            for (int nt = 0; nt < 4; ++nt) {
                const uint4* bp =
                    (const uint4*)&wt[(nt * 16 + l15) * WTS + kk + lh * 16];
                bfr[nt].q[0] = bp[0];
                bfr[nt].q[1] = bp[1];
            }
#pragma unroll
            for (int nt = 0; nt < 4; ++nt)
                acc[nt] = wmma_bf16(a.v, bfr[nt].v, acc[nt]);
        }

        // epilogue: C-layout -> bf16 (q,k row-major; v transposed [H][T])
        if (w < 2) {
            __bf16* dst = (w == 0) ? qb : kb;
#pragma unroll
            for (int nt = 0; nt < 4; ++nt) {
#pragma unroll
                for (int r = 0; r < 8; ++r) {
                    int m = rowbase + r + lh * 8;
                    dst[(size_t)m * H_ + nt * 16 + l15] = (__bf16)acc[nt][r];
                }
            }
        } else {
            // V^T: regs r, r+1 are adjacent t positions of one h-row -> pack b32
            int bb = rowbase >> 10;              // 128-row tile never spans batches
            int tbase = (rowbase & (T_ - 1)) + lh * 8;
#pragma unroll
            for (int nt = 0; nt < 4; ++nt) {
                __bf16* vrow = vtb + ((size_t)bb * H_ + nt * 16 + l15) * T_;
#pragma unroll
                for (int r = 0; r < 8; r += 2) {
                    union { __bf16 h[2]; uint32_t u; } pk;
                    pk.h[0] = (__bf16)acc[nt][r];
                    pk.h[1] = (__bf16)acc[nt][r + 1];
                    *(uint32_t*)&vrow[tbase + r] = pk.u;
                }
            }
        }
    }
}

// ---------------------------------------------------------------------------
// Kernel 2: flash attention. grid (T/128, B), 128 threads = 4 waves.
// Wave owns 32 query rows (2 M-tiles); 32-key K/V^T blocks double-buffered in
// LDS via async global->LDS loads (ASYNCcnt pipelining).
// ---------------------------------------------------------------------------
__global__ __launch_bounds__(128) void flash_attn(
    const __bf16* __restrict__ qb, const __bf16* __restrict__ kb,
    const __bf16* __restrict__ vtb, float* __restrict__ out)
{
    __shared__ __bf16 kbuf[2][32 * 64];      // [buf][s_local][h]
    __shared__ __bf16 vbuf[2][64 * 32];      // [buf][h][s_local]  (V^T)
    __shared__ __bf16 pbuf[4][2][16 * 32];   // wave-private P staging

    const int tid  = threadIdx.x;
    const int lane = tid & 31;
    const int wave = tid >> 5;
    const int l15  = lane & 15;
    const int lh   = lane >> 4;
    const int b    = blockIdx.y;
    const int qblk = blockIdx.x;                 // 0..7
    const int qwave = qblk * 128 + wave * 32;    // 32 rows per wave

    const __bf16* qbB = qb  + (size_t)b * T_ * H_;
    const __bf16* kbB = kb  + (size_t)b * T_ * H_;
    const __bf16* vtB = vtb + (size_t)b * H_ * T_;

    // Q A-fragments: 2 M-tiles x 2 K-steps
    Frag aq[2][2];
#pragma unroll
    for (int m = 0; m < 2; ++m) {
        const __bf16* qrow = qbB + (size_t)(qwave + m * 16 + l15) * H_;
#pragma unroll
        for (int s = 0; s < 2; ++s) {
            aq[m][s].q[0] = *(const uint4*)(qrow + s * 32 + lh * 16);
            aq[m][s].q[1] = *(const uint4*)(qrow + s * 32 + lh * 16 + 8);
        }
    }

    float m_[2][8], l_[2][8];
    v8f o[2][4];
#pragma unroll
    for (int m = 0; m < 2; ++m) {
#pragma unroll
        for (int r = 0; r < 8; ++r) { m_[m][r] = -__builtin_inff(); l_[m][r] = 0.f; }
#pragma unroll
        for (int nt = 0; nt < 4; ++nt) o[m][nt] = v8f_zero();
    }

    // async stage of one 32-key block (OPS_PER_BLOCK async ops per thread)
    auto stage = [&](int s0, int buf) {
#pragma unroll
        for (int j = 0; j < 2048 / (128 * CHUNK_ELEMS); ++j) {
            int e = (tid + j * 128) * CHUNK_ELEMS;
            int s = e >> 6, h = e & 63;
            async_copy_chunk(kbB + (size_t)(s0 + s) * H_ + h, &kbuf[buf][e]);
        }
#pragma unroll
        for (int j = 0; j < 2048 / (128 * CHUNK_ELEMS); ++j) {
            int e = (tid + j * 128) * CHUNK_ELEMS;
            int h = e >> 5, s = e & 31;
            async_copy_chunk(vtB + (size_t)h * T_ + s0 + s, &vbuf[buf][e]);
        }
    };

    const int nblk = 4 * qblk + 4;       // causal: keys up to qblk*128+127
    stage(0, 0);
    for (int kbI = 0; kbI < nblk; ++kbI) {
        const int s0  = kbI * 32;
        const int cur = kbI & 1;
        if (kbI + 1 < nblk) {            // prefetch next block into other buffer
            stage((kbI + 1) * 32, cur ^ 1);
            wait_asynccnt<OPS_PER_BLOCK>();  // newest outstanding = next block only
        } else {
            wait_asynccnt<0>();
        }
        __syncthreads();                 // current buffer visible to all waves

        if (s0 <= qwave + 31) {
            const __bf16* kd = kbuf[cur];
            const __bf16* vd = vbuf[cur];

            // ---- preload K fragments once; reused by both M-tiles ----
            Frag bk[2][2];
#pragma unroll
            for (int t = 0; t < 2; ++t) {
#pragma unroll
                for (int ks = 0; ks < 2; ++ks) {
                    const uint4* p =
                        (const uint4*)&kd[(t * 16 + l15) * 64 + ks * 32 + lh * 16];
                    bk[t][ks].q[0] = p[0];
                    bk[t][ks].q[1] = p[1];
                }
            }

#pragma unroll
            for (int m = 0; m < 2; ++m) {
                const int qbase = qwave + m * 16;
                if (s0 > qbase + 15) continue;   // sub-tile above diagonal

                // ---- S = Q K^T ----
                v8f sf[2];
#pragma unroll
                for (int t = 0; t < 2; ++t) {
                    v8f c = v8f_zero();
                    c = wmma_bf16(aq[m][0].v, bk[t][0].v, c);
                    c = wmma_bf16(aq[m][1].v, bk[t][1].v, c);
                    sf[t] = c;
                }
                // ---- scale + causal mask ----
#pragma unroll
                for (int t = 0; t < 2; ++t) {
                    int sg = s0 + t * 16 + l15;
#pragma unroll
                    for (int r = 0; r < 8; ++r) {
                        float v = sf[t][r] * SCALE_F;
                        int qr = qbase + r + lh * 8;
                        sf[t][r] = (sg > qr) ? -__builtin_inff() : v;
                    }
                }
                // ---- online softmax in registers ----
                __bf16* pl = pbuf[wave][m];
#pragma unroll
                for (int r = 0; r < 8; ++r) {
                    float mc = rowmax16(fmaxf(sf[0][r], sf[1][r]));
                    float mn = fmaxf(m_[m][r], mc);
                    float corr = __expf(m_[m][r] - mn);
                    m_[m][r] = mn;
                    float p0 = __expf(sf[0][r] - mn);
                    float p1 = __expf(sf[1][r] - mn);
                    l_[m][r] = l_[m][r] * corr + rowsum16(p0 + p1);
#pragma unroll
                    for (int nt = 0; nt < 4; ++nt) o[m][nt][r] *= corr;
                    int row = r + lh * 8;
                    pl[row * 32 + l15]      = (__bf16)p0;
                    pl[row * 32 + 16 + l15] = (__bf16)p1;
                }
            }
            asm volatile("s_wait_dscnt 0" ::: "memory");

            // ---- preload V fragments once; reused by both M-tiles ----
            Frag bv[4];
#pragma unroll
            for (int nt = 0; nt < 4; ++nt) {
                const uint4* vp =
                    (const uint4*)&vd[(nt * 16 + l15) * 32 + lh * 16];
                bv[nt].q[0] = vp[0];
                bv[nt].q[1] = vp[1];
            }
            // ---- O += P V ----
#pragma unroll
            for (int m = 0; m < 2; ++m) {
                if (s0 > qwave + m * 16 + 15) continue;
                const __bf16* pl = pbuf[wave][m];
                Frag ap;
                ap.q[0] = *(const uint4*)&pl[l15 * 32 + lh * 16];
                ap.q[1] = *(const uint4*)&pl[l15 * 32 + lh * 16 + 8];
#pragma unroll
                for (int nt = 0; nt < 4; ++nt)
                    o[m][nt] = wmma_bf16(ap.v, bv[nt].v, o[m][nt]);
            }
        }
        __syncthreads();   // everyone done with buf[cur] before it is re-staged
    }

    // ---- epilogue: O / l, fp32 out [B][T][H] ----
#pragma unroll
    for (int m = 0; m < 2; ++m) {
        float inv[8];
#pragma unroll
        for (int r = 0; r < 8; ++r) inv[r] = 1.f / l_[m][r];
#pragma unroll
        for (int nt = 0; nt < 4; ++nt) {
#pragma unroll
            for (int r = 0; r < 8; ++r) {
                int row = qwave + m * 16 + r + lh * 8;
                __builtin_nontemporal_store(
                    o[m][nt][r] * inv[r],
                    &out[((size_t)b * T_ + row) * H_ + nt * 16 + l15]);
            }
        }
    }
}

extern "C" void kernel_launch(void* const* d_in, const int* in_sizes, int n_in,
                              void* d_out, int out_size, void* d_ws, size_t ws_size,
                              hipStream_t stream) {
    (void)in_sizes; (void)n_in; (void)out_size; (void)ws_size;
    const float* x  = (const float*)d_in[0];
    const float* Wq = (const float*)d_in[1];
    const float* Wk = (const float*)d_in[2];
    const float* Wv = (const float*)d_in[3];
    float* out = (float*)d_out;

    char* ws = (char*)d_ws;
    __bf16* qb  = (__bf16*)(ws);
    __bf16* kb  = (__bf16*)(ws + (size_t)4 * 1024 * 1024);
    __bf16* vtb = (__bf16*)(ws + (size_t)8 * 1024 * 1024);

    proj_qkv<<<dim3(B_ * T_ / 128), dim3(256), 0, stream>>>(x, Wq, Wk, Wv,
                                                            qb, kb, vtb);
    flash_attn<<<dim3(T_ / 128, B_), dim3(128), 0, stream>>>(qb, kb, vtb, out);
}
